// MMD_loss3_73967926772315
// MI455X (gfx1250) — matrix operand order repaired and checked by reading, not compiled, not run on hardware
//
#include <hip/hip_runtime.h>
#include <hip/hip_bf16.h>

// ---------------- types ----------------
typedef __bf16  v8bf  __attribute__((ext_vector_type(8)));
typedef __bf16  v16bf __attribute__((ext_vector_type(16)));
typedef float   v8f   __attribute__((ext_vector_type(8)));

#define NCH   21
#define D     4096
#define BROW  128
#define NROW  256          // 2B
#define LDSK  32           // K chunk per iteration (one bf16 WMMA K)
#define LDS_STRIDE 40      // bf16 elems per LDS row (80B, 16B-aligned, bank-spread)

// row gi of "total" for channel c:  rows 0..127 = source, 128..255 = target
__device__ __forceinline__ const float* row_ptr(const float* __restrict__ s,
                                                const float* __restrict__ t,
                                                int c, int gi) {
    const float* base = (gi < BROW) ? s : t;
    int r = (gi < BROW) ? gi : gi - BROW;
    return base + ((size_t)r * NCH + (size_t)c) * D;
}

__device__ __forceinline__ float block_reduce_256(float v, float* buf) {
    int t = threadIdx.x;
    buf[t] = v; __syncthreads();
    for (int s = 128; s > 0; s >>= 1) {
        if (t < s) buf[t] += buf[t + s];
        __syncthreads();
    }
    float r = buf[0]; __syncthreads();
    return r;
}

// ---------------- 1) row squared norms (fp32, matches reference sq) -----------
__global__ void sq_kernel(const float* __restrict__ src, const float* __restrict__ tgt,
                          float* __restrict__ sqw) {
    __shared__ float buf[256];
    int c  = blockIdx.x;
    int gi = blockIdx.y;
    const float* p = row_ptr(src, tgt, c, gi);
    float s = 0.f;
    for (int k = threadIdx.x; k < D; k += 256) { float x = p[k]; s += x * x; }
    float r = block_reduce_256(s, buf);
    if (threadIdx.x == 0) sqw[c * NROW + gi] = r;
}

// ---------------- 2) Gram via split-bf16 WMMA ---------------------------------
// grid (21, 4, 4): workgroup computes 64x64 tile (i0,j0) of G_c = X X^T.
__global__ void __launch_bounds__(256)
gram_kernel(const float* __restrict__ src, const float* __restrict__ tgt,
            float* __restrict__ gram) {
    __shared__ __bf16 a_hi[64 * LDS_STRIDE];
    __shared__ __bf16 a_lo[64 * LDS_STRIDE];
    __shared__ __bf16 b_hi[64 * LDS_STRIDE];
    __shared__ __bf16 b_lo[64 * LDS_STRIDE];

    const int c  = blockIdx.x;
    const int i0 = blockIdx.y * 64;
    const int j0 = blockIdx.z * 64;

    const int tid  = threadIdx.x;
    const int wave = tid >> 5;
    const int lane = tid & 31;
    const int half = lane >> 4;     // 0: lanes 0-15, 1: lanes 16-31
    const int m    = lane & 15;

    const int ti  = wave >> 1;        // 0..3  (16-row tile within 64)
    const int tjb = (wave & 1) * 2;   // 0 or 2 (this wave owns tiles tjb, tjb+1)

    // loader mapping: 256 threads cover 64 rows x 32 floats (8 floats/thread)
    const int lr = tid >> 2;          // 0..63
    const int lk = (tid & 3) * 8;     // 0,8,16,24

    const float* arow = row_ptr(src, tgt, c, i0 + lr);
    const float* brow = row_ptr(src, tgt, c, j0 + lr);

    v8f acc0 = {}; // tile (ti, tjb)
    v8f acc1 = {}; // tile (ti, tjb+1)

    for (int k0 = 0; k0 < D; k0 += LDSK) {
        // ---- global fp32 -> split bf16 hi/lo -> LDS ----
        {
            const float4* pa = (const float4*)(arow + k0 + lk);
            const float4* pb = (const float4*)(brow + k0 + lk);
            float xa[8], xb[8];
            float4 a0 = pa[0], a1 = pa[1], b0 = pb[0], b1 = pb[1];
            xa[0]=a0.x; xa[1]=a0.y; xa[2]=a0.z; xa[3]=a0.w;
            xa[4]=a1.x; xa[5]=a1.y; xa[6]=a1.z; xa[7]=a1.w;
            xb[0]=b0.x; xb[1]=b0.y; xb[2]=b0.z; xb[3]=b0.w;
            xb[4]=b1.x; xb[5]=b1.y; xb[6]=b1.z; xb[7]=b1.w;
            int idx = lr * LDS_STRIDE + lk;
            #pragma unroll
            for (int e = 0; e < 8; ++e) {
                __bf16 h = (__bf16)xa[e];
                a_hi[idx + e] = h;
                a_lo[idx + e] = (__bf16)(xa[e] - (float)h);
                __bf16 g = (__bf16)xb[e];
                b_hi[idx + e] = g;
                b_lo[idx + e] = (__bf16)(xb[e] - (float)g);
            }
        }
        __syncthreads();

        // ---- assemble fragments (contiguous k order, same for A and B) ----
        // A 16x32 bf16: lane half=0 -> K0..15; half=1 -> K16..31
        const int ar = ti * 16 + m;
        v8bf ah0 = *(const v8bf*)&a_hi[ar * LDS_STRIDE + half * 16];
        v8bf ah1 = *(const v8bf*)&a_hi[ar * LDS_STRIDE + half * 16 + 8];
        v8bf al0 = *(const v8bf*)&a_lo[ar * LDS_STRIDE + half * 16];
        v8bf al1 = *(const v8bf*)&a_lo[ar * LDS_STRIDE + half * 16 + 8];
        v16bf afh = __builtin_shufflevector(ah0, ah1, 0,1,2,3,4,5,6,7,8,9,10,11,12,13,14,15);
        v16bf afl = __builtin_shufflevector(al0, al1, 0,1,2,3,4,5,6,7,8,9,10,11,12,13,14,15);

        // B 32x16 bf16 (col n = row j0+tj*16+n of X): half=0 -> K0..15, half=1 -> K16..31
        const int br0 = tjb * 16 + m;
        const int br1 = br0 + 16;
        v8bf bh00 = *(const v8bf*)&b_hi[br0 * LDS_STRIDE + half * 16];
        v8bf bh01 = *(const v8bf*)&b_hi[br0 * LDS_STRIDE + half * 16 + 8];
        v8bf bl00 = *(const v8bf*)&b_lo[br0 * LDS_STRIDE + half * 16];
        v8bf bl01 = *(const v8bf*)&b_lo[br0 * LDS_STRIDE + half * 16 + 8];
        v8bf bh10 = *(const v8bf*)&b_hi[br1 * LDS_STRIDE + half * 16];
        v8bf bh11 = *(const v8bf*)&b_hi[br1 * LDS_STRIDE + half * 16 + 8];
        v8bf bl10 = *(const v8bf*)&b_lo[br1 * LDS_STRIDE + half * 16];
        v8bf bl11 = *(const v8bf*)&b_lo[br1 * LDS_STRIDE + half * 16 + 8];
        v16bf bfh0 = __builtin_shufflevector(bh00, bh01, 0,1,2,3,4,5,6,7,8,9,10,11,12,13,14,15);
        v16bf bfl0 = __builtin_shufflevector(bl00, bl01, 0,1,2,3,4,5,6,7,8,9,10,11,12,13,14,15);
        v16bf bfh1 = __builtin_shufflevector(bh10, bh11, 0,1,2,3,4,5,6,7,8,9,10,11,12,13,14,15);
        v16bf bfl1 = __builtin_shufflevector(bl10, bl11, 0,1,2,3,4,5,6,7,8,9,10,11,12,13,14,15);

        // ---- split-precision accumulate: hi*hi + hi*lo + lo*hi ----
        acc0 = __builtin_amdgcn_wmma_f32_16x16x32_bf16(false, afh, false, bfh0, (short)0, acc0, false, false);
        acc0 = __builtin_amdgcn_wmma_f32_16x16x32_bf16(false, afh, false, bfl0, (short)0, acc0, false, false);
        acc0 = __builtin_amdgcn_wmma_f32_16x16x32_bf16(false, afl, false, bfh0, (short)0, acc0, false, false);
        acc1 = __builtin_amdgcn_wmma_f32_16x16x32_bf16(false, afh, false, bfh1, (short)0, acc1, false, false);
        acc1 = __builtin_amdgcn_wmma_f32_16x16x32_bf16(false, afh, false, bfl1, (short)0, acc1, false, false);
        acc1 = __builtin_amdgcn_wmma_f32_16x16x32_bf16(false, afl, false, bfh1, (short)0, acc1, false, false);

        __syncthreads();
    }

    // ---- store C/D: VGPR r holds (M = r + half*8, N = m) ----
    float* gbase = gram + (size_t)c * (NROW * NROW);
    const int orow = i0 + ti * 16 + half * 8;
    const int oc0  = j0 + tjb * 16 + m;
    #pragma unroll
    for (int r = 0; r < 8; ++r) {
        gbase[(size_t)(orow + r) * NROW + oc0]      = acc0[r];
        gbase[(size_t)(orow + r) * NROW + oc0 + 16] = acc1[r];
    }
}

// ---------------- 3) bandwidth per channel ------------------------------------
// sum(L2) = 2n*sum(sq) - 2*sum(G);  bw0 = sumL2/(n^2-n)/KERNEL_MUL^(5//2)
__global__ void bw_kernel(const float* __restrict__ gram, const float* __restrict__ sqw,
                          float* __restrict__ bww) {
    __shared__ float buf[256];
    int c = blockIdx.x;
    const float* g = gram + (size_t)c * (NROW * NROW);
    float s = 0.f;
    for (int idx = threadIdx.x; idx < NROW * NROW; idx += 256) s += g[idx];
    float sumG = block_reduce_256(s, buf);
    float sumSq = block_reduce_256(sqw[c * NROW + threadIdx.x], buf);
    if (threadIdx.x == 0) {
        float n = (float)NROW;
        float sumL2 = 2.f * n * sumSq - 2.f * sumG;
        bww[c] = sumL2 / (n * n - n) / 4.f;   // KERNEL_MUL^(KERNEL_NUM//2) = 4
    }
}

// ---------------- 4) signed multi-bandwidth kernel sum ------------------------
__global__ void mmd_kernel(const float* __restrict__ gram, const float* __restrict__ sqw,
                           const float* __restrict__ bww, float* __restrict__ chw) {
    __shared__ float buf[256];
    int c = blockIdx.x;
    const float* g  = gram + (size_t)c * (NROW * NROW);
    const float* sq = sqw + c * NROW;
    float bw0 = bww[c];
    float acc = 0.f;
    for (int idx = threadIdx.x; idx < BROW * BROW; idx += 256) {
        int i = idx >> 7, j = idx & 127;
        float si = sq[i], sj = sq[j], sI = sq[BROW + i], sJ = sq[BROW + j];
        float lxx = si + sj - 2.f * g[(size_t)i * NROW + j];
        float lyy = sI + sJ - 2.f * g[(size_t)(BROW + i) * NROW + (BROW + j)];
        float lxy = si + sJ - 2.f * g[(size_t)i * NROW + (BROW + j)];
        float lyx = sI + sj - 2.f * g[(size_t)(BROW + i) * NROW + j];
        float bw = bw0;
        #pragma unroll
        for (int k = 0; k < 5; ++k) {
            float inv = -1.f / bw;
            acc += __expf(lxx * inv) + __expf(lyy * inv)
                 - __expf(lxy * inv) - __expf(lyx * inv);
            bw *= 2.f;   // KERNEL_MUL
        }
    }
    float r = block_reduce_256(acc, buf);
    if (threadIdx.x == 0) chw[c] = r / (float)(BROW * BROW);
}

// ---------------- 5) final fixed-order sum over channels ----------------------
__global__ void final_kernel(const float* __restrict__ chw, float* __restrict__ out) {
    if (threadIdx.x == 0) {
        float s = 0.f;
        for (int c = 0; c < NCH; ++c) s += chw[c];
        out[0] = s / (float)NCH;
    }
}

// ---------------- launch ------------------------------------------------------
extern "C" void kernel_launch(void* const* d_in, const int* in_sizes, int n_in,
                              void* d_out, int out_size, void* d_ws, size_t ws_size,
                              hipStream_t stream) {
    const float* src = (const float*)d_in[0];
    const float* tgt = (const float*)d_in[1];
    float* ws   = (float*)d_ws;
    float* gram = ws;                                  // 21*256*256
    float* sqw  = gram + (size_t)NCH * NROW * NROW;    // 21*256
    float* bww  = sqw + NCH * NROW;                    // 21
    float* chw  = bww + NCH;                           // 21
    float* out  = (float*)d_out;

    sq_kernel  <<<dim3(NCH, NROW), 256, 0, stream>>>(src, tgt, sqw);
    gram_kernel<<<dim3(NCH, 4, 4), 256, 0, stream>>>(src, tgt, gram);
    bw_kernel  <<<NCH, 256, 0, stream>>>(gram, sqw, bww);
    mmd_kernel <<<NCH, 256, 0, stream>>>(gram, sqw, bww, chw);
    final_kernel<<<1, 32, 0, stream>>>(chw, out);
}
